// PatternMemoryDTW_8143257993997
// MI455X (gfx1250) — matrix operand order repaired and checked by reading, not compiled, not run on hardware
//
#include <hip/hip_runtime.h>
#include <hip/hip_bf16.h>
#include <math.h>

typedef __attribute__((ext_vector_type(16))) _Float16 v16h;
typedef __attribute__((ext_vector_type(8)))  float    v8f;

#define INF_ 1000000000.0f

// ---------------------------------------------------------------------------
// Kernel A: per node n (one workgroup, 8 waves):
//   - stage nodes[n] (16x128 f32) in LDS, qsq per row
//   - each wave handles 8 prototypes m: 4x v_wmma_f32_16x16x32_f16 (K=128)
//   - cost = sqrt(max(q2 + p2 - 2*dot, 1e-12)) -> LDS tile
//   - 16-lane anti-diagonal DTW scan -> dists[n][m]
// ---------------------------------------------------------------------------
__global__ __launch_bounds__(256) void cost_dtw_kernel(
    const float* __restrict__ nodes,       // [N,16,128]
    const float* __restrict__ protos,      // [M,16,128]
    float* __restrict__ dists,             // [N,M]
    int N, int M)
{
    constexpr int D  = 128;
    constexpr int Kq = 16;
    constexpr int AP = 132;                 // padded row stride (bank-conflict free)

    __shared__ __align__(16) float Asub[Kq * AP];     // 8.25 KB
    __shared__ float qsq[Kq];
    __shared__ float ctile[8 * 16 * 17];              // per-wave 16x16(+1 pad) cost

    const int tid  = threadIdx.x;
    const int wave = tid >> 5;
    const int lane = tid & 31;
    const int n    = blockIdx.x;

    // ---- stage node tile ----
    {
        const float* gp = nodes + (size_t)n * Kq * D;
        for (int idx = tid; idx < Kq * D; idx += 256)
            Asub[(idx >> 7) * AP + (idx & 127)] = gp[idx];
    }
    __syncthreads();
    if (tid < Kq) {
        float s = 0.f;
        const float* ar = &Asub[tid * AP];
        for (int k = 0; k < D; ++k) s += ar[k] * ar[k];
        qsq[tid] = s;
    }
    __syncthreads();

    // ---- per-lane A fragments (constant across all m handled by this wave) ----
    const int arow = lane & 15;
    const int koff = (lane >> 4) * 16;
    v16h afrag[4];
#pragma unroll
    for (int c = 0; c < 4; ++c) {
        const float* ap0 = &Asub[arow * AP + c * 32 + koff];
        const float* ap1 = ap0 + 8;
        v16h a;
#pragma unroll
        for (int e = 0; e < 8; ++e) {
            a[e]     = (_Float16)ap0[e];
            a[e + 8] = (_Float16)ap1[e];
        }
        afrag[c] = a;
    }

    const int col   = lane & 15;
    const int khalf = (lane >> 4) * 16;
    const int rbase = (lane >> 4) * 8;
    float* myc = &ctile[wave * (16 * 17)];

    for (int mi = 0; mi < 8; ++mi) {
        const int m = wave * 8 + mi;

        // ---- B fragments + WMMA accumulate over K=128 ----
        v8f acc = {0.f, 0.f, 0.f, 0.f, 0.f, 0.f, 0.f, 0.f};
        float p2p = 0.f;
        const float* pb = protos + ((size_t)m * 16 + col) * D + khalf;
#pragma unroll
        for (int c = 0; c < 4; ++c) {
            const float4* bp = (const float4*)(pb + c * 32);
            float4 q0 = bp[0], q1 = bp[1], q2 = bp[2], q3 = bp[3];
            v16h b;
            b[0]=(_Float16)q0.x;  b[1]=(_Float16)q0.y;  b[2]=(_Float16)q0.z;  b[3]=(_Float16)q0.w;
            b[4]=(_Float16)q1.x;  b[5]=(_Float16)q1.y;  b[6]=(_Float16)q1.z;  b[7]=(_Float16)q1.w;
            b[8]=(_Float16)q2.x;  b[9]=(_Float16)q2.y;  b[10]=(_Float16)q2.z; b[11]=(_Float16)q2.w;
            b[12]=(_Float16)q3.x; b[13]=(_Float16)q3.y; b[14]=(_Float16)q3.z; b[15]=(_Float16)q3.w;
            p2p += q0.x*q0.x + q0.y*q0.y + q0.z*q0.z + q0.w*q0.w
                 + q1.x*q1.x + q1.y*q1.y + q1.z*q1.z + q1.w*q1.w
                 + q2.x*q2.x + q2.y*q2.y + q2.z*q2.z + q2.w*q2.w
                 + q3.x*q3.x + q3.y*q3.y + q3.z*q3.z + q3.w*q3.w;
            acc = __builtin_amdgcn_wmma_f32_16x16x32_f16(
                false, afrag[c], false, b, (short)0, acc, false, false);
        }
        const float p2 = p2p + __shfl_xor(p2p, 16, 32);   // full sum over D for this column

        // ---- cost tile -> LDS (row-major, padded) ----
#pragma unroll
        for (int r = 0; r < 8; ++r) {
            const int rowc = rbase + r;
            float d2 = qsq[rowc] + p2 - 2.0f * acc[r];
            myc[rowc * 17 + col] = sqrtf(fmaxf(d2, 1e-12f));
        }
        __syncthreads();

        // ---- DTW: anti-diagonal systolic scan, lane i owns row i ----
        {
            const int i = lane & 15;
            const float* crow = &myc[i * 17];
            float cur = INF_, left = INF_, prev2 = INF_;
            for (int t = 0; t < 31; ++t) {
                const float s = __shfl_up(cur, 1, 32);  // lane i-1's value, step t-1
                const int j = t - i;
                if (j >= 0 && j < 16) {
                    const float cc = crow[j];
                    const float up = (i == 0) ? INF_ : s;
                    const float dg = (i == 0) ? ((j == 0) ? 0.0f : INF_) : prev2;
                    const float lf = (j == 0) ? INF_ : left;
                    cur = cc + fminf(fminf(up, lf), dg);
                    left = cur;
                } else {
                    cur = INF_;
                }
                prev2 = s;                               // lane i-1's value, step t-1 -> (t+1)-2
            }
            if (lane == 15) dists[(size_t)n * M + m] = cur;   // dp[15][15]
        }
        __syncthreads();
    }
}

// ---------------------------------------------------------------------------
// Kernel B: 16 nodes per workgroup.
//   softmax over M -> read = w @ memory_values ; summary = mean_k nodes
//   h = gelu_exact(fused @ w1 + b1) ; out = h @ w2 + b2
// ---------------------------------------------------------------------------
__global__ __launch_bounds__(256) void head_kernel(
    const float* __restrict__ nodes,   // [N,16,128]
    const float* __restrict__ memv,    // [64,128]
    const float* __restrict__ w1,      // [256,128]
    const float* __restrict__ b1,      // [128]
    const float* __restrict__ w2,      // [128,96]
    const float* __restrict__ b2,      // [96]
    const float* __restrict__ dists,   // [N,64]
    float* __restrict__ out,           // [N,96]
    int N)
{
    constexpr int NB = 16, M = 64, D = 128, P = 96;
    __shared__ float wts[NB * M];          //  4 KB
    __shared__ float mems[M * D];          // 32 KB
    __shared__ float fused[NB * 2 * D];    // 16 KB
    __shared__ float hbuf[NB * D];         //  8 KB

    const int tid = threadIdx.x;
    const int n0  = blockIdx.x * NB;

    for (int idx = tid; idx < M * D; idx += 256) mems[idx] = memv[idx];

    if (tid < NB) {                        // softmax over M per node
        const float* dr = dists + (size_t)(n0 + tid) * M;
        float mn = dr[0];
        for (int m = 1; m < M; ++m) mn = fminf(mn, dr[m]);
        float sum = 0.f;
        for (int m = 0; m < M; ++m) { float e = expf(mn - dr[m]); wts[tid * M + m] = e; sum += e; }
        const float inv = 1.0f / sum;
        for (int m = 0; m < M; ++m) wts[tid * M + m] *= inv;
    }
    __syncthreads();

    for (int idx = tid; idx < NB * D; idx += 256) {     // fused = [summary | read]
        const int r = idx >> 7, d = idx & 127;
        const float* np = nodes + ((size_t)(n0 + r) * 16) * D + d;
        float s = 0.f;
#pragma unroll
        for (int k = 0; k < 16; ++k) s += np[k * D];
        fused[r * 2 * D + d] = s * (1.0f / 16.0f);
        const float* wr = &wts[r * M];
        float rd = 0.f;
        for (int m = 0; m < M; ++m) rd += wr[m] * mems[m * D + d];
        fused[r * 2 * D + D + d] = rd;
    }
    __syncthreads();

    {                                                   // h = gelu(fused @ w1 + b1)
        const int j = tid & 127, rg = tid >> 7;
        float accv[8];
        const float bj = b1[j];
#pragma unroll
        for (int r = 0; r < 8; ++r) accv[r] = bj;
        for (int i = 0; i < 2 * D; ++i) {
            const float wv = w1[i * D + j];
#pragma unroll
            for (int r = 0; r < 8; ++r)
                accv[r] += fused[(rg * 8 + r) * 2 * D + i] * wv;
        }
#pragma unroll
        for (int r = 0; r < 8; ++r) {
            const float x = accv[r];
            hbuf[(rg * 8 + r) * D + j] = 0.5f * x * (1.0f + erff(x * 0.70710678118654752f));
        }
    }
    __syncthreads();

    for (int q = 0; q < 6; ++q) {                       // out = h @ w2 + b2
        const int idx = tid * 6 + q;                    // NB*P = 1536 = 256*6
        const int row = idx / P, p = idx - row * P;
        float accv = b2[p];
        const float* hr = &hbuf[row * D];
        for (int j = 0; j < D; ++j) accv += hr[j] * w2[j * P + p];
        out[(size_t)(n0 + row) * P + p] = accv;
    }
}

extern "C" void kernel_launch(void* const* d_in, const int* in_sizes, int n_in,
                              void* d_out, int out_size, void* d_ws, size_t ws_size,
                              hipStream_t stream) {
    const float* nodes  = (const float*)d_in[0];
    const float* protos = (const float*)d_in[1];
    const float* memv   = (const float*)d_in[2];
    const float* w1     = (const float*)d_in[3];
    const float* b1     = (const float*)d_in[4];
    const float* w2     = (const float*)d_in[5];
    const float* b2     = (const float*)d_in[6];

    const int D = in_sizes[4];               // 128
    const int P = in_sizes[6];               // 96
    const int M = in_sizes[2] / D;           // 64
    const int N = in_sizes[0] / (16 * D);    // 4096

    float* out   = (float*)d_out;
    float* dists = out + (size_t)N * P;      // tuple output: [N,P] then [N,M]

    cost_dtw_kernel<<<N, 256, 0, stream>>>(nodes, protos, dists, N, M);
    head_kernel<<<N / 16, 256, 0, stream>>>(nodes, memv, w1, b1, w2, b2, dists, out, N);
}
